// VQVAE_conv_47072841564927
// MI455X (gfx1250) — matrix operand order repaired and checked by reading, not compiled, not run on hardware
//
#include <hip/hip_runtime.h>

typedef __bf16 bf16;
typedef __attribute__((ext_vector_type(16))) __bf16 bf16x16;
typedef __attribute__((ext_vector_type(8)))  float  f32x8;

#define NSAMP  16384
#define MT     32          // samples per workgroup (2 WMMA row tiles)
#define NWAVES 8
#define BLOCK  256

// ---- d_out element offsets (flat concat of tuple outputs) ----
#define OUT_ZQ    0ull          // (N,512,1)
#define OUT_XHAT  8388608ull    // (N,256,10)
#define OUT_IDX   50331648ull   // (N,1)
#define OUT_EMB   50348032ull   // emb, qq, ac scalars

// ---- workspace byte offsets ----
#define OFF_W1   0u        // 128x256 bf16
#define OFF_W2   65536u    // 256x512 bf16
#define OFF_W3   327680u   // 512x1024 bf16
#define OFF_W4   1376256u  // 512x512 bf16
#define OFF_CB   1900544u  // 1024x512 bf16
#define OFF_WD1  2949120u  // 512x512 bf16
#define OFF_WD2  3473408u  // 1024x512 bf16
#define OFF_WD3  4521984u  // 1280x1024 bf16 (expanded transposed conv)
#define OFF_WD4  7143424u  // 256x128 bf16
#define OFF_CN   7208960u  // 1024 f32 codebook norms
#define OFF_PART 7213056u  // 512 f32 loss partials

// =================== weight prep kernels ===================
__global__ void k_cast(const float* __restrict__ s, bf16* __restrict__ d, int n){
  int i = blockIdx.x*BLOCK + threadIdx.x; if (i < n) d[i] = (bf16)s[i];
}
__global__ void k_w2(const float* __restrict__ ew2, bf16* __restrict__ W2){ // [co][tap*128+ci]
  int i = blockIdx.x*BLOCK + threadIdx.x;
  int co = i>>9, q = i&511, tap = q>>7, ci = q&127;
  W2[i] = (bf16)ew2[(((co<<7)+ci)<<2) + tap];
}
__global__ void k_w3(const float* __restrict__ ew3, bf16* __restrict__ W3){ // [co][p*256+ci]
  int i = blockIdx.x*BLOCK + threadIdx.x;
  int co = i>>10, q = i&1023, p = q>>8, ci = q&255;
  W3[i] = (bf16)ew3[(((co<<8)+ci)<<2) + p];
}
__global__ void k_wd1(const float* __restrict__ dw1, bf16* __restrict__ WD1){ // transpose 512x512
  int i = blockIdx.x*BLOCK + threadIdx.x;
  int co = i>>9, ci = i&511;
  WD1[i] = (bf16)dw1[(ci<<9) + co];
}
__global__ void k_wd2(const float* __restrict__ dw2, bf16* __restrict__ WD2){ // [(j*256+c)][ci]
  int i = blockIdx.x*BLOCK + threadIdx.x;
  int f = i>>9, ci = i&511, j = f>>8, c = f&255;
  WD2[i] = (bf16)dw2[(((ci<<8)+c)<<2) + j];
}
__global__ void k_wd3(const float* __restrict__ dw3, bf16* __restrict__ WD3){ // [(j*128+co)][(i*256+ci)]
  int i = blockIdx.x*BLOCK + threadIdx.x;
  int fo = i>>10, g = i&1023;
  int j = fo>>7, co = fo&127, ii = g>>8, ci = g&255;
  int t = j - 2*ii;
  float v = (t >= 0 && t < 4) ? dw3[(((ci<<7)+co)<<2) + t] : 0.0f;
  WD3[i] = (bf16)v;
}
__global__ void k_wd4(const float* __restrict__ dw4, bf16* __restrict__ WD4){ // transpose 128x256
  int i = blockIdx.x*BLOCK + threadIdx.x;
  int co = i>>7, ci = i&127;
  WD4[i] = (bf16)dw4[(ci<<8) + co];
}
__global__ void k_cnorm(const float* __restrict__ cb, float* __restrict__ cn){
  int e = blockIdx.x*BLOCK + threadIdx.x; if (e >= 1024) return;
  const float* r = cb + (size_t)e*512; float s = 0.f;
  for (int d = 0; d < 512; ++d) s += r[d]*r[d];
  cn[e] = s;
}
__global__ void k_final(const float* __restrict__ part, float* __restrict__ out){
  if (threadIdx.x == 0){
    float s = 0.f;
    for (int i = 0; i < NSAMP/MT; ++i) s += part[i];
    out[OUT_EMB]   = s * (1.25f / ((float)NSAMP * 512.f));
    out[OUT_EMB+1] = 0.f;
    out[OUT_EMB+2] = 0.f;
  }
}

// =================== WMMA fragment loaders ===================
union FragU { bf16x16 v; uint4 q[2]; };

// A 16x32 bf16 fragment from row-major [row][K] (ISA 7.12.2 16-bit A layout):
// lanes 0-15 -> rows 0-15 hold K 0..7 & 16..23 ; lanes 16-31 hold K 8..15 & 24..31
__device__ __forceinline__ bf16x16 ldA(const bf16* base, int stride, int lane){
  int r  = lane & 15;
  int k0 = (lane >> 4) << 3;
  const bf16* p = base + r*stride + k0;
  FragU f;
  f.q[0] = *(const uint4*)(p);
  f.q[1] = *(const uint4*)(p + 16);
  return f.v;
}
// B 32x16 bf16 fragment from W stored [col][K] (transposed so K runs contiguous):
// lanes 0-15 -> cols, K 0..7 & 16..23 ; lanes 16-31 -> cols, K 8..15 & 24..31
__device__ __forceinline__ bf16x16 ldB(const bf16* __restrict__ W, int K, int ct, int k0, int lane){
  const bf16* p = W + (size_t)(ct + (lane & 15))*K + k0 + ((lane >> 4) << 3);
  FragU f;
  f.q[0] = *(const uint4*)(p);
  f.q[1] = *(const uint4*)(p + 16);
  return f.v;
}

// Generic 32xNOUT = (32xK) @ (KxNOUT) + bias [+relu], A in LDS bf16, W global bf16 [NOUT][K]
__device__ __forceinline__ void gemm_stage(
    const bf16* A, int As, const bf16* __restrict__ W, int K, int NOUT,
    const float* __restrict__ bias, int bmask, bool relu,
    bf16* outB, int outS, float* outF, int outFS,
    float* gout, long grs, int gcs,
    int lane, int wave)
{
  for (int ct = wave*16; ct < NOUT; ct += NWAVES*16){
    f32x8 acc0 = {0,0,0,0,0,0,0,0};
    f32x8 acc1 = {0,0,0,0,0,0,0,0};
    for (int k0 = 0; k0 < K; k0 += 32){
      bf16x16 b  = ldB(W, K, ct, k0, lane);
      bf16x16 a0 = ldA(A + k0, As, lane);
      bf16x16 a1 = ldA(A + 16*As + k0, As, lane);
      acc0 = __builtin_amdgcn_wmma_f32_16x16x32_bf16(false, a0, false, b, (short)0, acc0, false, false);
      acc1 = __builtin_amdgcn_wmma_f32_16x16x32_bf16(false, a1, false, b, (short)0, acc1, false, false);
    }
    int col = ct + (lane & 15);
    float bv = bias ? bias[col & bmask] : 0.0f;
    int rb = (lane >> 4) << 3;           // C layout: VGPR v -> row v (+8 for hi lanes)
    #pragma unroll
    for (int v = 0; v < 8; ++v){
      float x0 = acc0[v] + bv, x1 = acc1[v] + bv;
      if (relu){ x0 = fmaxf(x0, 0.f); x1 = fmaxf(x1, 0.f); }
      int r0 = rb + v, r1 = 16 + rb + v;
      if (outB){ outB[r0*outS + col] = (bf16)x0; outB[r1*outS + col] = (bf16)x1; }
      if (outF){ outF[r0*outFS + col] = x0;      outF[r1*outFS + col] = x1; }
      if (gout){ gout[(long)r0*grs + (long)col*gcs] = x0;
                 gout[(long)r1*grs + (long)col*gcs] = x1; }
    }
  }
}

__device__ __forceinline__ unsigned fkey(float f){ // monotonic u32 key for float min
  unsigned u = __float_as_uint(f);
  return (u & 0x80000000u) ? ~u : (u | 0x80000000u);
}

// VQ scoring: per column-tile compute d = |c|^2 - 2*(z.c); merge (key,idx) via ds_min_u64
__device__ __forceinline__ void vq_stage(
    const bf16* Zb, const bf16* __restrict__ CB, const float* __restrict__ cn,
    unsigned long long* mins, int lane, int wave)
{
  for (int ct = wave*16; ct < 1024; ct += NWAVES*16){
    f32x8 acc0 = {0,0,0,0,0,0,0,0};
    f32x8 acc1 = {0,0,0,0,0,0,0,0};
    for (int k0 = 0; k0 < 512; k0 += 32){
      bf16x16 b  = ldB(CB, 512, ct, k0, lane);
      bf16x16 a0 = ldA(Zb + k0, 512, lane);
      bf16x16 a1 = ldA(Zb + 16*512 + k0, 512, lane);
      acc0 = __builtin_amdgcn_wmma_f32_16x16x32_bf16(false, a0, false, b, (short)0, acc0, false, false);
      acc1 = __builtin_amdgcn_wmma_f32_16x16x32_bf16(false, a1, false, b, (short)0, acc1, false, false);
    }
    int col = ct + (lane & 15);
    float c2 = cn[col];
    int rb = (lane >> 4) << 3;
    #pragma unroll
    for (int v = 0; v < 8; ++v){
      float d0 = c2 - 2.0f*acc0[v];
      float d1 = c2 - 2.0f*acc1[v];
      atomicMin(&mins[rb + v],      ((unsigned long long)fkey(d0) << 32) | (unsigned)col);
      atomicMin(&mins[16 + rb + v], ((unsigned long long)fkey(d1) << 32) | (unsigned)col);
    }
  }
}

// =================== fused VQ-VAE kernel ===================
__launch_bounds__(BLOCK, 1)
__global__ void k_vqvae(
    const float* __restrict__ x,  const float* __restrict__ cbf,
    const float* __restrict__ eb1, const float* __restrict__ eb2,
    const float* __restrict__ eb3, const float* __restrict__ eb4,
    const float* __restrict__ db1, const float* __restrict__ db2,
    const float* __restrict__ db3, const float* __restrict__ db4,
    const bf16* __restrict__ W1,  const bf16* __restrict__ W2,
    const bf16* __restrict__ W3,  const bf16* __restrict__ W4,
    const bf16* __restrict__ CB,  const bf16* __restrict__ WD1,
    const bf16* __restrict__ WD2, const bf16* __restrict__ WD3,
    const bf16* __restrict__ WD4, const float* __restrict__ cn,
    float* __restrict__ out, float* __restrict__ part)
{
  __shared__ __align__(16) unsigned char smem[198016]; // aliased regions, peak ~193KB
  const int tid  = threadIdx.x;
  const int lane = tid & 31;
  const int wave = tid >> 5;
  const int n0   = blockIdx.x * MT;

  // ---- encoder conv1 (k=1): per position t, 256 -> 128 ----
  bf16* XAT = (bf16*)(smem + 0);        // 32x256 bf16
  bf16* F1  = (bf16*)(smem + 16384);    // 32x1280 bf16, f = t*128 + co
  for (int t = 0; t < 10; ++t){
    __syncthreads();
    for (int e = tid; e < MT*256; e += BLOCK){
      int r = e >> 8, ci = e & 255;
      XAT[(r << 8) + ci] = (bf16)x[(size_t)(n0 + r)*2560 + ci*10 + t];
    }
    __syncthreads();
    gemm_stage(XAT, 256, W1, 256, 128, eb1, 127, true,
               F1 + t*128, 1280, nullptr, 0, nullptr, 0, 0, lane, wave);
  }
  // ---- encoder conv2 (k=4,s=2): window p*256..+512 -> 256 ----
  bf16* F2 = (bf16*)(smem + 98304);     // 32x1024, f = p*256 + co
  __syncthreads();
  for (int p = 0; p < 4; ++p)
    gemm_stage(F1 + p*256, 1280, W2, 512, 256, eb2, 255, true,
               F2 + p*256, 1024, nullptr, 0, nullptr, 0, 0, lane, wave);
  // ---- encoder conv3 (k=4,s=2): 1024 -> 512 ----
  bf16* F3 = (bf16*)(smem + 0);         // 32x512
  __syncthreads();
  gemm_stage(F2, 1024, W3, 1024, 512, eb3, 511, true,
             F3, 512, nullptr, 0, nullptr, 0, 0, lane, wave);
  // ---- encoder conv4 (k=1): 512 -> 512, keep f32 (z) and bf16 (VQ A) ----
  float* ZF = (float*)(smem + 32768);   // 32x512 f32
  bf16*  ZB = (bf16*)(smem + 163840);   // 32x512 bf16
  __syncthreads();
  gemm_stage(F3, 512, W4, 512, 512, eb4, 511, false,
             ZB, 512, ZF, 512, nullptr, 0, 0, lane, wave);
  // ---- VQ: argmin_e ( |c_e|^2 - 2 z.c_e ) via packed ds_min_u64 ----
  unsigned long long* MINS = (unsigned long long*)(smem + 196608);
  __syncthreads();
  if (tid < MT) MINS[tid] = ~0ull;
  __syncthreads();
  vq_stage(ZB, CB, cn, MINS, lane, wave);
  __syncthreads();
  int*   IDX = (int*)(smem + 196864);
  float* RED = (float*)(smem + 196992);
  if (tid < MT){
    int id = (int)(MINS[tid] & 0xffffffffull);
    IDX[tid] = id;
    out[OUT_IDX + (size_t)n0 + tid] = (float)id;
  }
  __syncthreads();
  // ---- gather z_q (f32 codebook), emit zq3 output, loss partial, bf16 for dec ----
  bf16* ZQB = (bf16*)(smem + 98304);    // 32x512
  float lsum = 0.f;
  for (int e = tid; e < MT*512; e += BLOCK){
    int r = e >> 9, c = e & 511;
    float q  = cbf[(size_t)IDX[r]*512 + c];
    float zv = ZF[(r << 9) + c];
    ZQB[(r << 9) + c] = (bf16)q;
    out[OUT_ZQ + (size_t)(n0 + r)*512 + c] = q;
    float dq = q - zv; lsum += dq*dq;
  }
  RED[tid] = lsum;
  __syncthreads();
  if (tid == 0){
    float s = 0.f;
    for (int i = 0; i < BLOCK; ++i) s += RED[i];
    part[blockIdx.x] = s;   // deterministic: fixed-order finalize kernel sums these
  }
  // ---- decoder convT1 (k=1): 512 -> 512 ----
  bf16* G1 = (bf16*)(smem + 0);
  gemm_stage(ZQB, 512, WD1, 512, 512, db1, 511, true,
             G1, 512, nullptr, 0, nullptr, 0, 0, lane, wave);
  // ---- decoder convT2 (k=4,s=2): 512 -> 4x256 ----
  bf16* G2 = (bf16*)(smem + 32768);     // 32x1024, f = j*256 + c
  __syncthreads();
  gemm_stage(G1, 512, WD2, 512, 1024, db2, 255, true,
             G2, 1024, nullptr, 0, nullptr, 0, 0, lane, wave);
  // ---- decoder convT3 (k=4,s=2): 4x256 -> 10x128 (expanded weights) ----
  bf16* G3 = (bf16*)(smem + 98304);     // 32x1280, f = j*128 + c
  __syncthreads();
  gemm_stage(G2, 1024, WD3, 1024, 1280, db3, 127, true,
             G3, 1280, nullptr, 0, nullptr, 0, 0, lane, wave);
  // ---- decoder convT4 (k=1): per position j, 128 -> 256, write x_hat ----
  __syncthreads();
  for (int j = 0; j < 10; ++j)
    gemm_stage(G3 + j*128, 1280, WD4, 128, 256, db4, 255, false,
               nullptr, 0, nullptr, 0,
               out + OUT_XHAT + (size_t)n0*2560 + j, 2560, 10, lane, wave);
}

// =================== launcher ===================
extern "C" void kernel_launch(void* const* d_in, const int* in_sizes, int n_in,
                              void* d_out, int out_size, void* d_ws, size_t ws_size,
                              hipStream_t stream)
{
  const float* x   = (const float*)d_in[0];
  const float* cbf = (const float*)d_in[1];
  const float* ew1 = (const float*)d_in[2];  const float* eb1 = (const float*)d_in[3];
  const float* ew2 = (const float*)d_in[4];  const float* eb2 = (const float*)d_in[5];
  const float* ew3 = (const float*)d_in[6];  const float* eb3 = (const float*)d_in[7];
  const float* ew4 = (const float*)d_in[8];  const float* eb4 = (const float*)d_in[9];
  const float* dw1 = (const float*)d_in[10]; const float* db1 = (const float*)d_in[11];
  const float* dw2 = (const float*)d_in[12]; const float* db2 = (const float*)d_in[13];
  const float* dw3 = (const float*)d_in[14]; const float* db3 = (const float*)d_in[15];
  const float* dw4 = (const float*)d_in[16]; const float* db4 = (const float*)d_in[17];
  float* out = (float*)d_out;
  char*  ws  = (char*)d_ws;

  bf16*  W1  = (bf16*)(ws + OFF_W1);
  bf16*  W2  = (bf16*)(ws + OFF_W2);
  bf16*  W3  = (bf16*)(ws + OFF_W3);
  bf16*  W4  = (bf16*)(ws + OFF_W4);
  bf16*  CBw = (bf16*)(ws + OFF_CB);
  bf16*  WD1 = (bf16*)(ws + OFF_WD1);
  bf16*  WD2 = (bf16*)(ws + OFF_WD2);
  bf16*  WD3 = (bf16*)(ws + OFF_WD3);
  bf16*  WD4 = (bf16*)(ws + OFF_WD4);
  float* cn  = (float*)(ws + OFF_CN);
  float* part= (float*)(ws + OFF_PART);

  k_cast <<<  32768/BLOCK, BLOCK, 0, stream>>>(ew1, W1, 32768);
  k_w2   <<< 131072/BLOCK, BLOCK, 0, stream>>>(ew2, W2);
  k_w3   <<< 524288/BLOCK, BLOCK, 0, stream>>>(ew3, W3);
  k_cast <<< 262144/BLOCK, BLOCK, 0, stream>>>(ew4, W4, 262144);
  k_cast <<< 524288/BLOCK, BLOCK, 0, stream>>>(cbf, CBw, 524288);
  k_wd1  <<< 262144/BLOCK, BLOCK, 0, stream>>>(dw1, WD1);
  k_wd2  <<< 524288/BLOCK, BLOCK, 0, stream>>>(dw2, WD2);
  k_wd3  <<<1310720/BLOCK, BLOCK, 0, stream>>>(dw3, WD3);
  k_wd4  <<<  32768/BLOCK, BLOCK, 0, stream>>>(dw4, WD4);
  k_cnorm<<<4, BLOCK, 0, stream>>>(cbf, cn);

  k_vqvae<<<NSAMP/MT, BLOCK, 0, stream>>>(x, cbf, eb1, eb2, eb3, eb4,
                                          db1, db2, db3, db4,
                                          W1, W2, W3, W4, CBw,
                                          WD1, WD2, WD3, WD4, cn, out, part);
  k_final<<<1, 64, 0, stream>>>(part, out);

  (void)in_sizes; (void)n_in; (void)out_size; (void)ws_size;
}
